// GCN_26714696581619
// MI455X (gfx1250) — matrix operand order repaired and checked by reading, not compiled, not run on hardware
//
#include <hip/hip_runtime.h>
#include <hip/hip_bf16.h>

typedef __bf16 bf16;
typedef __attribute__((ext_vector_type(16))) __bf16 v16bf;
typedef __attribute__((ext_vector_type(8)))  __bf16 v8bf;
typedef __attribute__((ext_vector_type(8)))  float  v8f;

// ---------------- small elementwise / prep kernels ----------------

__global__ void k_init_deg(float* __restrict__ deg, int N) {
    int i = blockIdx.x * blockDim.x + threadIdx.x;
    if (i < N) deg[i] = 1.0f;              // self-loop weight
}

__global__ void k_accum_deg(const long long* __restrict__ ei,
                            const float* __restrict__ ew,
                            float* __restrict__ deg, int E) {
    int e = blockIdx.x * blockDim.x + threadIdx.x;
    if (e < E) atomicAdd(&deg[(int)ei[(size_t)E + e]], ew[e]);
}

__global__ void k_dinv(float* __restrict__ deg, int N) {
    int i = blockIdx.x * blockDim.x + threadIdx.x;
    if (i < N) {
        float v = deg[i];
        deg[i] = (v > 0.0f) ? rsqrtf(v) : 0.0f;
    }
}

__global__ void k_norm(const long long* __restrict__ ei,
                       const float* __restrict__ ew,
                       const float* __restrict__ dinv,
                       float* __restrict__ nrm, int E, int N) {
    int e = blockIdx.x * blockDim.x + threadIdx.x;
    if (e >= E + N) return;
    int r, c; float w;
    if (e < E) { r = (int)ei[e]; c = (int)ei[(size_t)E + e]; w = ew[e]; }
    else       { r = c = e - E; w = 1.0f; }
    nrm[e] = dinv[r] * w * dinv[c];
}

__global__ void k_to_bf16(const float* __restrict__ in, bf16* __restrict__ out, int total) {
    int i = blockIdx.x * blockDim.x + threadIdx.x;
    if (i < total) out[i] = (bf16)in[i];
}

// W[K][N] f32 (row-major)  ->  Wt[N][K] bf16 (transposed, row-major)
__global__ void k_w_transpose_bf16(const float* __restrict__ W, bf16* __restrict__ Wt,
                                   int K, int Nd) {
    int i = blockIdx.x * blockDim.x + threadIdx.x;
    if (i >= K * Nd) return;
    int k = i / Nd, n = i % Nd;
    Wt[(size_t)n * K + k] = (bf16)W[i];
}

__global__ void k_bias_relu_bf16(const float* __restrict__ agg, const float* __restrict__ b,
                                 bf16* __restrict__ out, int total, int mask) {
    int i = blockIdx.x * blockDim.x + threadIdx.x;
    if (i >= total) return;
    float v = agg[i] + b[i & mask];
    out[i] = (bf16)fmaxf(v, 0.0f);
}

__global__ void k_bias_add(float* __restrict__ out, const float* __restrict__ b,
                           int total, int mask) {
    int i = blockIdx.x * blockDim.x + threadIdx.x;
    if (i < total) out[i] += b[i & mask];
}

// ---------------- WMMA bf16 GEMM:  C[M][ND] = A[M][KD] @ Bt[ND][KD]^T ----------------
// One wave owns one 16-wide N-tile; B fragments live in registers for the whole
// grid-stride loop over M tiles. Layouts follow CDNA5 ISA 7.12.2 (wave32).
// All A-fragment loads for an M-tile are issued before the WMMA chain so the
// backend can use partial s_wait_loadcnt and overlap loads with WMMAs.

template <int KD, int ND>
__global__ void k_wmma_gemm(const bf16* __restrict__ A, const bf16* __restrict__ Bt,
                            float* __restrict__ C, int M) {
    const int lane = threadIdx.x & 31;
    const int wave = threadIdx.x >> 5;      // n-tile index within block
    const int half = lane >> 4;             // 0: lanes 0-15, 1: lanes 16-31
    const int lm   = lane & 15;
    const int n0   = wave * 16;

    constexpr int KT = KD / 32;             // 16x16x32 steps along K
    v16bf bfrag[KT];
#pragma unroll
    for (int kt = 0; kt < KT; ++kt) {
        // B lane layout: N=lm; elems 0-7 = K+half*8..+7, elems 8-15 = K+16+half*8..+7
        const bf16* bp = Bt + (size_t)(n0 + lm) * KD + kt * 32 + half * 8;
        v8bf blo = *(const v8bf*)(bp), bhi = *(const v8bf*)(bp + 16);
#pragma unroll
        for (int i = 0; i < 8; ++i) { bfrag[kt][i] = blo[i]; bfrag[kt][i + 8] = bhi[i]; }
    }

    const int mtiles = M >> 4;
    for (int mt = blockIdx.x; mt < mtiles; mt += gridDim.x) {
        const int m0 = mt << 4;
        const bf16* ar = A + (size_t)(m0 + lm) * KD + half * 8;

        // issue ALL A loads for this tile first (independent b128s, one clause)
        v8bf lo[KT], hi[KT];
#pragma unroll
        for (int kt = 0; kt < KT; ++kt) {
            lo[kt] = *(const v8bf*)(ar + kt * 32);
            hi[kt] = *(const v8bf*)(ar + kt * 32 + 16);
        }

        v8f acc = {};
#pragma unroll
        for (int kt = 0; kt < KT; ++kt) {
            // A lane layout (16-bit 16x32): row M=lm; elems 0-7 = K+half*8..+7,
            // elems 8-15 = K+16+half*8..+7
            v16bf a;
#pragma unroll
            for (int i = 0; i < 8; ++i) { a[i] = lo[kt][i]; a[i + 8] = hi[kt][i]; }
            acc = __builtin_amdgcn_wmma_f32_16x16x32_bf16(
                false, a, false, bfrag[kt], (short)0, acc, false, false);
        }

        // D layout: VGPR v -> row m0 + half*8 + v, col n0 + lm
        float* crow = C + (size_t)(m0 + half * 8) * ND + n0 + lm;
#pragma unroll
        for (int v = 0; v < 8; ++v) crow[(size_t)v * ND] = acc[v];
    }
}

// ---------------- edge gather * norm -> scatter-add (atomics resolve in L2) -------------

template <int F>
__global__ void k_aggregate(const float* __restrict__ xw, const long long* __restrict__ ei,
                            const float* __restrict__ nrm, float* __restrict__ agg,
                            int E, int N) {
    constexpr unsigned TPE = F / 4;         // threads per edge, power of two
    unsigned tid = blockIdx.x * blockDim.x + threadIdx.x;
    unsigned total = (unsigned)(E + N) * TPE;
    if (tid >= total) return;
    unsigned e = tid / TPE;
    int f = (int)(tid % TPE) * 4;
    int r, c;
    if (e < (unsigned)E) { r = (int)ei[e]; c = (int)ei[(size_t)E + e]; }
    else                 { r = c = (int)(e - (unsigned)E); }
    float nm = nrm[e];
    const float4 v = *(const float4*)(xw + (size_t)r * F + f);
    float* dst = agg + (size_t)c * F + f;
    atomicAdd(dst + 0, v.x * nm);
    atomicAdd(dst + 1, v.y * nm);
    atomicAdd(dst + 2, v.z * nm);
    atomicAdd(dst + 3, v.w * nm);
}

// ---------------------------------- launcher ----------------------------------

extern "C" void kernel_launch(void* const* d_in, const int* in_sizes, int n_in,
                              void* d_out, int out_size, void* d_ws, size_t ws_size,
                              hipStream_t stream) {
    (void)n_in; (void)ws_size; (void)out_size;

    const float*     x  = (const float*)d_in[0];
    const long long* ei = (const long long*)d_in[1];
    const float*     ew = (const float*)d_in[2];
    const float*     W1 = (const float*)d_in[3];
    const float*     b1 = (const float*)d_in[4];
    const float*     W2 = (const float*)d_in[5];
    const float*     b2 = (const float*)d_in[6];
    float*           out = (float*)d_out;

    const int DIN = 128, DHID = 128, DOUT = 64;
    const int N = in_sizes[0] / DIN;
    const int E = in_sizes[2];

    // workspace carve-up (256B aligned)
    char* ws = (char*)d_ws;
    size_t off = 0;
    auto alloc = [&](size_t bytes) -> void* {
        void* p = ws + off;
        off += (bytes + 255) & ~(size_t)255;
        return p;
    };
    bf16*  xb   = (bf16*)alloc((size_t)N * DIN * sizeof(bf16));    // x bf16, reused for h bf16
    bf16*  W1t  = (bf16*)alloc((size_t)DIN * DHID * sizeof(bf16));
    bf16*  W2t  = (bf16*)alloc((size_t)DHID * DOUT * sizeof(bf16));
    float* xw   = (float*)alloc((size_t)N * DHID * sizeof(float)); // xw1, reused for xw2
    float* agg1 = (float*)alloc((size_t)N * DHID * sizeof(float));
    float* deg  = (float*)alloc((size_t)N * sizeof(float));        // deg -> dinv in place
    float* nrm  = (float*)alloc((size_t)(E + N) * sizeof(float));

    const int T = 256;
    auto nb = [](long long work, int t) { return (unsigned)((work + t - 1) / t); };

    // ---- normalization coefficients ----
    k_init_deg <<<nb(N, T), T, 0, stream>>>(deg, N);
    k_accum_deg<<<nb(E, T), T, 0, stream>>>(ei, ew, deg, E);
    k_dinv     <<<nb(N, T), T, 0, stream>>>(deg, N);
    k_norm     <<<nb(E + N, T), T, 0, stream>>>(ei, ew, deg, nrm, E, N);

    // ---- precision conversion ----
    k_to_bf16          <<<nb((long long)N * DIN, T), T, 0, stream>>>(x, xb, N * DIN);
    k_w_transpose_bf16 <<<nb(DIN * DHID, T), T, 0, stream>>>(W1, W1t, DIN, DHID);
    k_w_transpose_bf16 <<<nb(DHID * DOUT, T), T, 0, stream>>>(W2, W2t, DHID, DOUT);

    // ---- layer 1: xw = x @ W1 (WMMA bf16), aggregate, bias+relu -> bf16 h ----
    k_wmma_gemm<128, 128><<<1024, 256, 0, stream>>>(xb, W1t, xw, N);

    hipMemsetAsync(agg1, 0, (size_t)N * DHID * sizeof(float), stream);
    k_aggregate<128><<<nb((long long)(E + N) * 32, T), T, 0, stream>>>(xw, ei, nrm, agg1, E, N);
    k_bias_relu_bf16<<<nb((long long)N * DHID, T), T, 0, stream>>>(agg1, b1, xb, N * DHID, DHID - 1);

    // ---- layer 2: xw2 = h @ W2 (WMMA bf16), aggregate into d_out, bias ----
    k_wmma_gemm<128, 64><<<1024, 128, 0, stream>>>(xb, W2t, xw, N);

    hipMemsetAsync(out, 0, (size_t)N * DOUT * sizeof(float), stream);
    k_aggregate<64><<<nb((long long)(E + N) * 16, T), T, 0, stream>>>(xw, ei, nrm, out, E, N);
    k_bias_add<<<nb((long long)N * DOUT, T), T, 0, stream>>>(out, b2, N * DOUT, DOUT - 1);
}
